// VDProjAttention_82351702933750
// MI455X (gfx1250) — matrix operand order repaired and checked by reading, compiled, not run on hardware
//
#include <hip/hip_runtime.h>

typedef unsigned int u32;
typedef unsigned short u16;
typedef __attribute__((ext_vector_type(16))) __bf16 v16bf;
typedef __attribute__((ext_vector_type(8)))  float  v8f;
typedef __attribute__((ext_vector_type(4)))  int    v4i;

// ---------------- problem constants ----------------
constexpr int Bc = 2, Sc = 2048, Hc = 2048, NHc = 16, HDc = 128;
constexpr int VPc = 320, VHDc = 20, VPADc = 32;
constexpr float SCALEc = 0.08838834764831845f;   // 1/sqrt(128)

// ---------------- workspace layout (bytes) ----------------
constexpr size_t SZ_XB  = (size_t)Bc * Sc * Hc * 2;        // X bf16 [M][K]
constexpr size_t SZ_WQB = (size_t)Hc * Hc * 2;             // Wq/Wk bf16 transposed [N][K]
constexpr size_t SZ_WVB = (size_t)Hc * VPc * 2;            // Wv bf16 transposed
constexpr size_t SZ_WDB = (size_t)VPc * Hc * 2;            // Wd bf16 transposed
constexpr size_t SZ_QB  = (size_t)Bc * NHc * Sc * HDc * 2; // Q heads bf16 [bh][s][128]
constexpr size_t SZ_VB  = (size_t)Bc * NHc * Sc * VPADc * 2; // V dim-major [bh][32][S]
constexpr size_t SZ_AB  = (size_t)Bc * Sc * VPc * 2;       // attn out bf16

constexpr size_t OFF_XB  = 0;
constexpr size_t OFF_WQB = OFF_XB  + SZ_XB;
constexpr size_t OFF_WKB = OFF_WQB + SZ_WQB;
constexpr size_t OFF_WVB = OFF_WKB + SZ_WQB;
constexpr size_t OFF_WDB = OFF_WVB + SZ_WVB;
constexpr size_t OFF_QB  = OFF_WDB + SZ_WDB;
constexpr size_t OFF_KB  = OFF_QB  + SZ_QB;
constexpr size_t OFF_VB  = OFF_KB  + SZ_QB;
constexpr size_t OFF_AB  = OFF_VB  + SZ_VB;

// ---------------- async global->LDS (probe-confirmed) ----------------
#if defined(__has_builtin)
#if __has_builtin(__builtin_amdgcn_global_load_async_to_lds_b128)
#define HAVE_ASYNC_LDS 1
#endif
#endif
#ifndef HAVE_ASYNC_LDS
#define HAVE_ASYNC_LDS 0
#endif

typedef __attribute__((address_space(1))) v4i* gptr_v4i;
typedef __attribute__((address_space(3))) v4i* lptr_v4i;

// copies 16 bytes (8 bf16 halves)
__device__ __forceinline__ void copy_b128_to_lds(const u16* gsrc, u16* ldst) {
#if HAVE_ASYNC_LDS
    __builtin_amdgcn_global_load_async_to_lds_b128(
        (gptr_v4i)(v4i*)gsrc, (lptr_v4i)(v4i*)ldst, 0, 0);
#else
    *(uint4*)ldst = *(const uint4*)gsrc;
#endif
}
__device__ __forceinline__ void wait_lds_copies() {
#if HAVE_ASYNC_LDS
#if __has_builtin(__builtin_amdgcn_s_wait_asynccnt)
    __builtin_amdgcn_s_wait_asynccnt(0);
#else
    asm volatile("s_wait_asynccnt 0x0" ::: "memory");
#endif
#endif
}

// ---------------- helpers ----------------
__device__ __forceinline__ u16 f2bf(float f) {
    u32 u = __float_as_uint(f);
    u32 r = (u + 0x7FFFu + ((u >> 16) & 1u)) >> 16;   // round-to-nearest-even
    return (u16)r;
}
__device__ __forceinline__ float bf2f(u16 h) {
    return __uint_as_float(((u32)h) << 16);
}

// CDNA5 16-bit A-matrix 16x32: lane L holds row M=L%16; VGPR g -> K pair:
__device__ __forceinline__ int amap(int g, int lane) {
    return ((g & 4) ? 16 : 0) + ((lane & 16) ? 8 : 0) + ((g & 3) << 1);
}
// 16-bit B-matrix 32x16: lane L holds col N=L%16; same K pairs as amap.
__device__ __forceinline__ int bmap(int g, int lane) {
    return ((g & 4) ? 16 : 0) + ((lane & 16) ? 8 : 0) + ((g & 3) << 1);
}

__device__ __forceinline__ v8f wmma_bf16(const u32 au[8], const u32 bu[8], v8f c) {
    union U { u32 u[8]; v16bf v; } A, Bv;
#pragma unroll
    for (int i = 0; i < 8; ++i) { A.u[i] = au[i]; Bv.u[i] = bu[i]; }
    return __builtin_amdgcn_wmma_f32_16x16x32_bf16(false, A.v, false, Bv.v,
                                                   (short)0, c, false, false);
}

// ---------------- f32 -> bf16 conversion ----------------
__global__ void cvt_bf16_kernel(const float* __restrict__ src,
                                u16* __restrict__ dst, int n) {
    int i = blockIdx.x * blockDim.x + threadIdx.x;
    if (i < n) dst[i] = f2bf(src[i]);
}

// f32 [K][N] -> bf16 transposed [N][K].  grid: (ceil(N/256), K)
__global__ void cvtT_bf16_kernel(const float* __restrict__ src,
                                 u16* __restrict__ dst, int K, int N) {
    int n = blockIdx.x * blockDim.x + threadIdx.x;
    int k = blockIdx.y;
    if (n < N) dst[(size_t)n * K + k] = f2bf(src[(size_t)k * N + n]);
}

// ---------------- tiled bf16 GEMM: C[M,N] = A[M,K] @ W[K,N] + bias ----------------
// WT is the transposed weight [N][K].  Block tile 128x64, BK=32, 8 waves,
// double-buffered async global->LDS for both tiles.
// MODE 0: f32 row-major.  MODE 1: bf16 head-major [bh][s][PAD_].
// MODE 2: bf16 dim-major  [bh][d][S]  (for V).
template <int MODE, int HD_, int PAD_>
__global__ __launch_bounds__(256) void gemm_bf16_kernel(
        const u16* __restrict__ A, const u16* __restrict__ WT,
        const float* __restrict__ bias, void* __restrict__ Cout,
        int M, int N, int K) {
    __shared__ u16 As[2][128 * 40];   // [row][k], stride 40 halves
    __shared__ u16 Bs[2][64 * 40];    // [n][k]

    const int tid  = threadIdx.x;
    const int wave = tid >> 5;
    const int lane = tid & 31;
    const int ml   = lane & 15;
    const int mBlk = blockIdx.y * 128;
    const int nBlk = blockIdx.x * 64;

    auto issue_tiles = [&](int kk, int buf) {
        // A tile: 128 rows x 32 halves -> 2 b128 per thread
        int row = tid >> 1, c0 = (tid & 1) * 16;
        const u16* asrc = A + (size_t)(mBlk + row) * K + kk + c0;
        copy_b128_to_lds(asrc,     &As[buf][row * 40 + c0]);
        copy_b128_to_lds(asrc + 8, &As[buf][row * 40 + c0 + 8]);
        // B tile from WT: 64 rows x 32 halves -> 1 b128 per thread
        int n = tid >> 2, k0 = (tid & 3) * 8;
        copy_b128_to_lds(WT + (size_t)(nBlk + n) * K + kk + k0,
                         &Bs[buf][n * 40 + k0]);
    };

    v8f acc[4];
#pragma unroll
    for (int s = 0; s < 4; ++s)
        acc[s] = (v8f){0.f, 0.f, 0.f, 0.f, 0.f, 0.f, 0.f, 0.f};

    issue_tiles(0, 0);
    int buf = 0;
    for (int kk = 0; kk < K; kk += 32) {
        wait_lds_copies();
        __syncthreads();
        if (kk + 32 < K) issue_tiles(kk + 32, buf ^ 1);   // prefetch next tile

        u32 af[8];
#pragma unroll
        for (int g = 0; g < 8; ++g)
            af[g] = *(const u32*)&As[buf][(wave * 16 + ml) * 40 + amap(g, lane)];
#pragma unroll
        for (int s = 0; s < 4; ++s) {
            u32 bf_[8];
#pragma unroll
            for (int g = 0; g < 8; ++g)
                bf_[g] = *(const u32*)&Bs[buf][(s * 16 + ml) * 40 + bmap(g, lane)];
            acc[s] = wmma_bf16(af, bf_, acc[s]);
        }
        buf ^= 1;
    }

    // hoisted bias (4 columns per lane)
    float bcol[4];
#pragma unroll
    for (int s = 0; s < 4; ++s) bcol[s] = bias[nBlk + s * 16 + ml];

    const int hi = (lane & 16) ? 8 : 0;
#pragma unroll
    for (int s = 0; s < 4; ++s) {
#pragma unroll
        for (int r = 0; r < 8; ++r) {
            int n = nBlk + s * 16 + ml;
            int m = mBlk + wave * 16 + r + hi;
            float v = acc[s][r] + bcol[s];
            if (MODE == 0) {
                ((float*)Cout)[(size_t)m * N + n] = v;
            } else if (MODE == 1) {
                int b_ = m / Sc, s_ = m % Sc;
                int nh = n / HD_, d = n % HD_;
                ((u16*)Cout)[((size_t)(b_ * NHc + nh) * Sc + s_) * PAD_ + d] = f2bf(v);
            } else {   // MODE 2: dim-major V
                int b_ = m / Sc, s_ = m % Sc;
                int nh = n / HD_, d = n % HD_;
                ((u16*)Cout)[((size_t)(b_ * NHc + nh) * PAD_ + d) * Sc + s_] = f2bf(v);
            }
        }
    }
}

// ---------------- RoPE (+ optional gain) on bf16 heads [B*NH, S, 128] ----------------
// 64 threads per row: j<16 rotates pair (j, j+16); j in 16..63 scales the
// pass-through dims 32..127.  gain = 1/sqrt(HD) for Q (folds softmax scale), 1 for K.
__global__ void rope_kernel(u16* __restrict__ X, float gain) {
    int idx = blockIdx.x * blockDim.x + threadIdx.x;   // B*NH*S*64 threads
    int j  = idx & 63;
    int s  = (idx >> 6) & (Sc - 1);
    int bh = idx >> 17;
    if (bh >= Bc * NHc) return;
    size_t base = ((size_t)bh * Sc + s) * HDc;
    if (j < 16) {
        float inv = __powf(10000.f, -(float)(2 * j) / 32.f);
        float fr = (float)s * inv;
        float sn, cs;
        __sincosf(fr, &sn, &cs);
        float x1 = bf2f(X[base + j]);
        float x2 = bf2f(X[base + j + 16]);
        X[base + j]      = f2bf((x1 * cs - x2 * sn) * gain);
        X[base + j + 16] = f2bf((x2 * cs + x1 * sn) * gain);
    } else {
        int d = j + 16;            // 32..79
        X[base + d]      = f2bf(bf2f(X[base + d]) * gain);
        X[base + d + 48] = f2bf(bf2f(X[base + d + 48]) * gain);   // 80..127
    }
}

// ---------------- flash attention ----------------
// grid: (S/128, B*NH); 256 threads = 8 waves; wave owns 16 q rows.
// K-step = 32 keys; QK^T: 8 wmma; PV: 2 wmma.  K/V tiles double-buffered async.
// Softmax scale is pre-folded into Q.  Waves skip tiles fully above their diagonal.
__global__ __launch_bounds__(256) void flash_kernel(
        const u16* __restrict__ Qb, const u16* __restrict__ Kb,
        const u16* __restrict__ VtG, u16* __restrict__ Ab) {
    __shared__ u16 Kt[2][32 * 136];  // [key][dim], stride 136 halves
    __shared__ u16 Vt[2][32 * 40];   // [dim][key]
    __shared__ u16 Pt[8][16 * 40];   // per-wave P scratch [m][key]

    const int tid  = threadIdx.x;
    const int wave = tid >> 5;
    const int lane = tid & 31;
    const int ml   = lane & 15;
    const int hi   = (lane & 16) ? 8 : 0;

    const int bh    = blockIdx.y;
    const int b_    = bh / NHc;
    const int h_    = bh % NHc;
    const int qtile = blockIdx.x;
    const int qbase = qtile * 128 + wave * 16;
    const int rowhi = qbase + 15;    // last q row owned by this wave

    const u16* Qrow = Qb  + (size_t)bh * Sc * HDc;
    const u16* Krow = Kb  + (size_t)bh * Sc * HDc;
    const u16* Vrow = VtG + (size_t)bh * VPADc * Sc;   // [dim][s]

    auto issue_tiles = [&](int kt, int buf) {
        // K tile: 32 keys x 128 dims -> 2 b128 per thread
        int key = tid >> 3, c0 = (tid & 7) * 16;
        const u16* ksrc = Krow + (size_t)(kt * 32 + key) * HDc + c0;
        copy_b128_to_lds(ksrc,     &Kt[buf][key * 136 + c0]);
        copy_b128_to_lds(ksrc + 8, &Kt[buf][key * 136 + c0 + 8]);
        // V tile: 32 dims x 32 keys (contiguous in keys) -> 1 b128, 128 threads
        if (tid < 128) {
            int dim = tid >> 2, k0 = (tid & 3) * 8;
            copy_b128_to_lds(Vrow + (size_t)dim * Sc + kt * 32 + k0,
                             &Vt[buf][dim * 40 + k0]);
        }
    };

    // Q fragments (16 rows x 128 dims -> 4 K-steps of 32), loaded once
    u32 qf[4][8];
#pragma unroll
    for (int f = 0; f < 4; ++f)
#pragma unroll
        for (int g = 0; g < 8; ++g)
            qf[f][g] = *(const u32*)(Qrow + (size_t)(qbase + ml) * HDc +
                                     f * 32 + amap(g, lane));

    v8f o0 = (v8f){0.f,0.f,0.f,0.f,0.f,0.f,0.f,0.f};
    v8f o1 = o0;
    float mrow[8], lrow[8];
#pragma unroll
    for (int r = 0; r < 8; ++r) { mrow[r] = -1e30f; lrow[r] = 0.f; }

    const int ktmax = (qtile * 128 + 127) >> 5;   // inclusive, block max
    issue_tiles(0, 0);
    int buf = 0;
    for (int kt = 0; kt <= ktmax; ++kt) {
        wait_lds_copies();
        __syncthreads();
        if (kt < ktmax) issue_tiles(kt + 1, buf ^ 1);   // prefetch next tile

        // wave-uniform skip: tiles entirely above this wave's diagonal
        // contribute alpha=1, sum=0, dO=0 -> nothing to do.
        if (kt * 32 <= rowhi) {
            // scores: S(16x32) = Q(16x128) @ K^T(128x32)
            v8f s0 = (v8f){0.f,0.f,0.f,0.f,0.f,0.f,0.f,0.f};
            v8f s1 = s0;
#pragma unroll
            for (int f = 0; f < 4; ++f) {
                u32 b0[8], b1[8];
#pragma unroll
                for (int g = 0; g < 8; ++g) {
                    int k = f * 32 + bmap(g, lane);
                    b0[g] = *(const u32*)&Kt[buf][ml * 136 + k];
                    b1[g] = *(const u32*)&Kt[buf][(16 + ml) * 136 + k];
                }
                s0 = wmma_bf16(qf[f], b0, s0);
                s1 = wmma_bf16(qf[f], b1, s1);
            }

            // online softmax + P -> LDS (bf16); scale already folded into Q
            const int key0 = kt * 32 + ml;
            const int key1 = key0 + 16;
#pragma unroll
            for (int r = 0; r < 8; ++r) {
                int qg = qbase + r + hi;
                float x0 = s0[r]; if (key0 > qg) x0 = -1e30f;
                float x1 = s1[r]; if (key1 > qg) x1 = -1e30f;
                float mx = fmaxf(x0, x1);
#pragma unroll
                for (int off = 1; off < 16; off <<= 1)
                    mx = fmaxf(mx, __shfl_xor(mx, off, 32));
                float mnew  = fmaxf(mrow[r], mx);
                float alpha = __expf(mrow[r] - mnew);
                float p0 = __expf(x0 - mnew);
                float p1 = __expf(x1 - mnew);
                float sm = p0 + p1;
#pragma unroll
                for (int off = 1; off < 16; off <<= 1)
                    sm += __shfl_xor(sm, off, 32);
                lrow[r] = lrow[r] * alpha + sm;
                mrow[r] = mnew;
                o0[r] *= alpha;
                o1[r] *= alpha;
                int mloc = r + hi;
                Pt[wave][mloc * 40 + ml]      = f2bf(p0);
                Pt[wave][mloc * 40 + 16 + ml] = f2bf(p1);
            }
            // per-wave LDS writes then reads: DS ops are in-order within a wave

            // O(16x32) += P(16x32) @ V(32x32)
            u32 pa[8], vb0[8], vb1[8];
#pragma unroll
            for (int g = 0; g < 8; ++g) {
                pa[g]  = *(const u32*)&Pt[wave][ml * 40 + amap(g, lane)];
                int k  = bmap(g, lane);
                vb0[g] = *(const u32*)&Vt[buf][ml * 40 + k];
                vb1[g] = *(const u32*)&Vt[buf][(16 + ml) * 40 + k];
            }
            o0 = wmma_bf16(pa, vb0, o0);
            o1 = wmma_bf16(pa, vb1, o1);
        }
        buf ^= 1;
    }

    // finalize: O /= l, store dims < 20 to Ab[(b*S+s)*320 + h*20 + d]
#pragma unroll
    for (int r = 0; r < 8; ++r) {
        float invl = 1.f / lrow[r];
        int s_ = qbase + r + hi;
        size_t rowoff = ((size_t)(b_ * Sc + s_)) * VPc + h_ * VHDc;
        int d0 = ml;                         // dims 0..15, all < 20
        Ab[rowoff + d0] = f2bf(o0[r] * invl);
        int d1 = 16 + ml;                    // dims 16..31, keep < 20
        if (d1 < VHDc) Ab[rowoff + d1] = f2bf(o1[r] * invl);
    }
}

// ---------------- host launch ----------------
extern "C" void kernel_launch(void* const* d_in, const int* in_sizes, int n_in,
                              void* d_out, int out_size, void* d_ws, size_t ws_size,
                              hipStream_t stream) {
    const float* hs = (const float*)d_in[0];
    const float* Wq = (const float*)d_in[1];
    const float* bq = (const float*)d_in[2];
    const float* Wk = (const float*)d_in[3];
    const float* bk = (const float*)d_in[4];
    const float* Wv = (const float*)d_in[5];
    const float* bv = (const float*)d_in[6];
    const float* Wd = (const float*)d_in[7];
    const float* bd = (const float*)d_in[8];

    char* ws  = (char*)d_ws;
    u16* Xb   = (u16*)(ws + OFF_XB);
    u16* Wqb  = (u16*)(ws + OFF_WQB);   // transposed [N][K]
    u16* Wkb  = (u16*)(ws + OFF_WKB);
    u16* Wvb  = (u16*)(ws + OFF_WVB);
    u16* Wdb  = (u16*)(ws + OFF_WDB);
    u16* Qb   = (u16*)(ws + OFF_QB);
    u16* Kb   = (u16*)(ws + OFF_KB);
    u16* Vb   = (u16*)(ws + OFF_VB);
    u16* Ab   = (u16*)(ws + OFF_AB);

    const int M = Bc * Sc;   // 4096

    cvt_bf16_kernel<<<(M * Hc + 255) / 256, 256, 0, stream>>>(hs, Xb, M * Hc);
    // weights: f32 [K][N] -> bf16 transposed [N][K]
    cvtT_bf16_kernel<<<dim3(Hc / 256, Hc), 256, 0, stream>>>(Wq, Wqb, Hc, Hc);
    cvtT_bf16_kernel<<<dim3(Hc / 256, Hc), 256, 0, stream>>>(Wk, Wkb, Hc, Hc);
    cvtT_bf16_kernel<<<dim3((VPc + 255) / 256, Hc), 256, 0, stream>>>(Wv, Wvb, Hc, VPc);
    cvtT_bf16_kernel<<<dim3(Hc / 256, VPc), 256, 0, stream>>>(Wd, Wdb, VPc, Hc);

    // Q = X@Wq + bq -> bf16 [B,NH,S,128]
    gemm_bf16_kernel<1, HDc, HDc><<<dim3(Hc / 64, M / 128), 256, 0, stream>>>(
        Xb, Wqb, bq, Qb, M, Hc, Hc);
    // K
    gemm_bf16_kernel<1, HDc, HDc><<<dim3(Hc / 64, M / 128), 256, 0, stream>>>(
        Xb, Wkb, bk, Kb, M, Hc, Hc);
    // V -> bf16 dim-major [B*NH, 32, S] (pad rows zeroed)
    (void)hipMemsetAsync(Vb, 0, SZ_VB, stream);
    gemm_bf16_kernel<2, VHDc, VPADc><<<dim3(VPc / 64, M / 128), 256, 0, stream>>>(
        Xb, Wvb, bv, Vb, M, VPc, Hc);

    // RoPE on Q (with softmax scale folded in) and K
    {
        int n = Bc * NHc * Sc * 64;
        rope_kernel<<<n / 256, 256, 0, stream>>>(Qb, SCALEc);
        rope_kernel<<<n / 256, 256, 0, stream>>>(Kb, 1.0f);
    }

    // causal flash attention -> Ab bf16 [B*S, 320]
    flash_kernel<<<dim3(Sc / 128, Bc * NHc), 256, 0, stream>>>(Qb, Kb, Vb, Ab);

    // out = Ab @ Wd + bd -> f32 d_out [4096, 2048]
    gemm_bf16_kernel<0, HDc, HDc><<<dim3(Hc / 64, M / 128), 256, 0, stream>>>(
        Ab, Wdb, bd, (void*)d_out, M, Hc, VPc);
}
